// MoE_4088808865951
// MI455X (gfx1250) — compile-verified, matrix operand order checked by reading, not run on hardware
//
#include <hip/hip_runtime.h>

// ---------------- problem constants ----------------
constexpr int S_  = 4096;   // tokens
constexpr int D_  = 1024;   // dim
constexpr int H_  = 2816;   // moe intermediate
constexpr int E_  = 16;     // experts
constexpr int NT_ = S_ * 2;         // 8192 dispatched rows (K=2)
constexpr int MAXTILES = 80;        // 8192/128 + 15 partial + pad

// ---------------- types ----------------
typedef __bf16 bfx2  __attribute__((ext_vector_type(2)));
typedef __bf16 bfx8  __attribute__((ext_vector_type(8)));
typedef __bf16 bfx16 __attribute__((ext_vector_type(16)));
typedef float  fx8   __attribute__((ext_vector_type(8)));

#define CAT16(lo,hi) __builtin_shufflevector(lo,hi,0,1,2,3,4,5,6,7,8,9,10,11,12,13,14,15)

// native fptrunc -> v_cvt_pk_bf16_f32 on gfx1250 (RNE)
__device__ __forceinline__ unsigned int pack2bf(float x, float y) {
  bfx2 v;
  v[0] = (__bf16)x;
  v[1] = (__bf16)y;
  return __builtin_bit_cast(unsigned int, v);
}

// async global->LDS 16B copy (CDNA5 GLOBAL_LOAD_ASYNC_TO_LDS_B128, GVS mode)
__device__ __forceinline__ void async_copy_b128(const void* saddr_base,
                                                unsigned voffset_bytes,
                                                void* lds_ptr) {
  unsigned lds_off = (unsigned)(size_t)lds_ptr;   // low 32 bits == LDS offset
  asm volatile("global_load_async_to_lds_b128 %0, %1, %2"
               :: "v"(lds_off), "v"(voffset_bytes), "s"(saddr_base)
               : "memory");
}
__device__ __forceinline__ void async_wait0() {
  asm volatile("s_wait_asynccnt 0x0" ::: "memory");
}

// ---------------- 1. init ----------------
__global__ void init_kernel(int* counts, int* cursors, int* ntiles) {
  int t = threadIdx.x;
  if (t < E_) { counts[t] = 0; cursors[t] = 0; }
  if (t == 0) ntiles[0] = 0;
}

// ---------------- 2. router: logits, softmax, top-2, renorm ----------------
__global__ __launch_bounds__(256) void router_kernel(
    const float* __restrict__ x, const float* __restrict__ gw,
    float* __restrict__ logits_out, float* __restrict__ topi_out,
    float* __restrict__ rw, int* __restrict__ flat_e, int* __restrict__ counts)
{
  int wid = threadIdx.x >> 5, lane = threadIdx.x & 31;
  int tok = blockIdx.x * 8 + wid;              // one wave32 per token
  const float* xr = x + (size_t)tok * D_;
  float acc[E_];
  #pragma unroll
  for (int e = 0; e < E_; e++) acc[e] = 0.f;
  for (int d = lane; d < D_; d += 32) {
    float xv = xr[d];
    #pragma unroll
    for (int e = 0; e < E_; e++) acc[e] += xv * gw[e * D_ + d];
  }
  #pragma unroll
  for (int e = 0; e < E_; e++) {
    #pragma unroll
    for (int off = 16; off >= 1; off >>= 1)
      acc[e] += __shfl_xor(acc[e], off, 32);   // wave32 butterfly reduce
  }
  if (lane == 0) {
    float mx = acc[0];
    #pragma unroll
    for (int e = 1; e < E_; e++) mx = fmaxf(mx, acc[e]);
    float p[E_]; float sum = 0.f;
    #pragma unroll
    for (int e = 0; e < E_; e++) { p[e] = __expf(acc[e] - mx); sum += p[e]; }
    float inv = 1.f / sum;
    #pragma unroll
    for (int e = 0; e < E_; e++) p[e] *= inv;
    int i0 = 0; float p0 = p[0];
    #pragma unroll
    for (int e = 1; e < E_; e++) if (p[e] > p0) { p0 = p[e]; i0 = e; }
    int i1 = -1; float p1 = -1.f;
    #pragma unroll
    for (int e = 0; e < E_; e++) if (e != i0 && p[e] > p1) { p1 = p[e]; i1 = e; }
    float wsum = p0 + p1;
    #pragma unroll
    for (int e = 0; e < E_; e++) logits_out[(size_t)tok * E_ + e] = acc[e];
    topi_out[tok * 2 + 0] = (float)i0;
    topi_out[tok * 2 + 1] = (float)i1;
    rw[tok * 2 + 0] = p0 / wsum;
    rw[tok * 2 + 1] = p1 / wsum;
    flat_e[tok * 2 + 0] = i0;
    flat_e[tok * 2 + 1] = i1;
    atomicAdd(&counts[i0], 1);
    atomicAdd(&counts[i1], 1);
  }
}

// ---------------- 3. scan + build tile table (device-side group sizes) ----
__global__ void scan_kernel(const int* counts, int* cursors,
                            int* tileE, int* tileRow0, int* tileRows, int* ntiles) {
  if (threadIdx.x != 0) return;
  int run = 0, t = 0;
  for (int e = 0; e < E_; e++) {
    int c = counts[e];
    cursors[e] = run;
    int nt = (c + 127) >> 7;
    for (int i = 0; i < nt; i++) {
      tileE[t] = e;
      tileRow0[t] = run + i * 128;
      int rem = c - i * 128;
      tileRows[t] = rem < 128 ? rem : 128;
      t++;
    }
    run += c;
  }
  ntiles[0] = t;
}

// ---------------- 4. scatter: assign dispatched row positions -------------
__global__ void scatter_kernel(const int* __restrict__ flat_e, int* cursors,
                               int* __restrict__ pos_of_flat, int* __restrict__ row_of_pos) {
  int i = blockIdx.x * blockDim.x + threadIdx.x;
  if (i >= NT_) return;
  int e = flat_e[i];
  int pos = atomicAdd(&cursors[e], 1);
  pos_of_flat[i] = pos;
  row_of_pos[pos] = i;
}

// ---------------- 5. gather: build grouped bf16 activation rows -----------
__global__ __launch_bounds__(128) void gather_kernel(
    const float* __restrict__ x, const int* __restrict__ row_of_pos,
    unsigned short* __restrict__ xs)
{
  int pos = blockIdx.x;
  int tok = row_of_pos[pos] >> 1;
  const float4* src = (const float4*)(x + (size_t)tok * D_);
  int t = threadIdx.x;                         // 128 threads x 8 elems = 1024
  float4 a = src[t * 2], b = src[t * 2 + 1];
  uint4 o;
  o.x = pack2bf(a.x, a.y);
  o.y = pack2bf(a.z, a.w);
  o.z = pack2bf(b.x, b.y);
  o.w = pack2bf(b.z, b.w);
  *(uint4*)(xs + (size_t)pos * D_ + t * 8) = o;
}

// ---------------- 6. fused gate+up GEMM with SwiGLU epilogue --------------
// Block tile 128(M) x 64(N), KT=32, double-buffered LDS, 8 waves (4Mx2N),
// each wave: 2x2 wmma tiles per matrix (g,u) sharing A fragments.
constexpr int KT  = 32;
constexpr int LDA = 40;   // LDS row stride (ushorts), 80B = 16B-aligned

__global__ __launch_bounds__(256) void gemm_gateup(
    const unsigned short* __restrict__ xs, const float* __restrict__ wg,
    const float* __restrict__ wu, unsigned short* __restrict__ hbuf,
    const int* __restrict__ tileE, const int* __restrict__ tileRow0,
    const int* __restrict__ tileRows, const int* __restrict__ ntiles)
{
  __shared__ unsigned short As[2][128 * LDA];  // 20 KB
  __shared__ unsigned short Bg[2][64 * LDA];   // 10 KB
  __shared__ unsigned short Bu[2][64 * LDA];   // 10 KB

  int tile = blockIdx.y;
  if (tile >= ntiles[0]) return;               // uniform per block
  int e = tileE[tile], row0 = tileRow0[tile], rows = tileRows[tile];
  int n0 = blockIdx.x * 64;

  int tid  = threadIdx.x;
  int lane = tid & 31, wid = tid >> 5;
  int wm = (wid & 3) * 32;                     // 4 waves along M
  int wn = (wid >> 2) * 32;                    // 2 waves along N
  const size_t wbase = (size_t)e * H_ * D_;

  fx8 cg[2][2], cu[2][2];
  #pragma unroll
  for (int i = 0; i < 2; i++)
    #pragma unroll
    for (int j = 0; j < 2; j++) { cg[i][j] = {}; cu[i][j] = {}; }

  float4 gReg[2], uReg[2];

  // A tile: async copy straight into LDS (bf16, no conversion needed)
  auto asyncA = [&](int k0, int buf) {
    #pragma unroll
    for (int p = 0; p < 2; p++) {              // 128x32 bf16, 16B chunks
      int c = tid + p * 256;
      int r = c >> 2, col = (c & 3) * 8;
      int rg = row0 + r; if (rg >= NT_) rg = NT_ - 1;
      unsigned voff = (unsigned)(((size_t)rg * D_ + k0 + col) * 2);
      async_copy_b128(xs, voff, &As[buf][r * LDA + col]);
    }
  };
  auto loadB = [&](int k0) {
    #pragma unroll
    for (int p = 0; p < 2; p++) {              // B: 64x32 fp32 -> bf16
      int c = tid + p * 256;
      int r = c >> 3, col = (c & 7) * 4;
      size_t off = wbase + (size_t)(n0 + r) * D_ + k0 + col;
      gReg[p] = *(const float4*)(wg + off);
      uReg[p] = *(const float4*)(wu + off);
      if (k0 + 2 * KT < D_) {                  // stream prefetch 2 stages ahead
        __builtin_prefetch(wg + off + 2 * KT, 0, 3);
        __builtin_prefetch(wu + off + 2 * KT, 0, 3);
      }
    }
  };
  auto storeB = [&](int buf) {
    #pragma unroll
    for (int p = 0; p < 2; p++) {
      int c = tid + p * 256;
      int r = c >> 3, col = (c & 7) * 4;
      uint2 vg, vu;
      vg.x = pack2bf(gReg[p].x, gReg[p].y);
      vg.y = pack2bf(gReg[p].z, gReg[p].w);
      vu.x = pack2bf(uReg[p].x, uReg[p].y);
      vu.y = pack2bf(uReg[p].z, uReg[p].w);
      *(uint2*)&Bg[buf][r * LDA + col] = vg;
      *(uint2*)&Bu[buf][r * LDA + col] = vu;
    }
  };
  auto doMath = [&](int buf) {
    int kb = (lane < 16) ? 0 : 8;              // wave32 A/B K-half split
    bfx16 aFrag[2];
    #pragma unroll
    for (int ms = 0; ms < 2; ms++) {
      int ar = (wm + ms * 16 + (lane & 15)) * LDA;
      bfx8 lo = *(const bfx8*)&As[buf][ar + kb];
      bfx8 hi = *(const bfx8*)&As[buf][ar + kb + 16];
      aFrag[ms] = CAT16(lo, hi);
    }
    #pragma unroll
    for (int ns = 0; ns < 2; ns++) {
      int br = (wn + ns * 16 + (lane & 15)) * LDA;
      bfx8 glo = *(const bfx8*)&Bg[buf][br + kb];
      bfx8 ghi = *(const bfx8*)&Bg[buf][br + kb + 16];
      bfx16 bG = CAT16(glo, ghi);
      bfx8 ulo = *(const bfx8*)&Bu[buf][br + kb];
      bfx8 uhi = *(const bfx8*)&Bu[buf][br + kb + 16];
      bfx16 bU = CAT16(ulo, uhi);
      #pragma unroll
      for (int ms = 0; ms < 2; ms++) {
        cg[ms][ns] = __builtin_amdgcn_wmma_f32_16x16x32_bf16(
            false, aFrag[ms], false, bG, (short)0, cg[ms][ns], false, false);
        cu[ms][ns] = __builtin_amdgcn_wmma_f32_16x16x32_bf16(
            false, aFrag[ms], false, bU, (short)0, cu[ms][ns], false, false);
      }
    }
  };

  asyncA(0, 0);
  loadB(0);
  storeB(0);
  async_wait0();
  __syncthreads();
  const int NIT = D_ / KT;                     // 32
  #pragma unroll 2
  for (int kt = 0; kt < NIT; kt++) {
    int buf = kt & 1;
    if (kt + 1 < NIT) { asyncA((kt + 1) * KT, buf ^ 1); loadB((kt + 1) * KT); }
    doMath(buf);
    if (kt + 1 < NIT) storeB(buf ^ 1);
    async_wait0();
    __syncthreads();
  }

  // SwiGLU epilogue: h = silu(g) * u -> bf16
  int lhi = (lane >> 4) << 3;                  // 0 or +8 rows (C layout)
  #pragma unroll
  for (int ms = 0; ms < 2; ms++)
    #pragma unroll
    for (int ns = 0; ns < 2; ns++)
      #pragma unroll
      for (int r = 0; r < 8; r++) {
        int ml = wm + ms * 16 + r + lhi;
        if (ml < rows) {
          float g = cg[ms][ns][r], u = cu[ms][ns][r];
          float hv = g * u / (1.f + __expf(-g));
          __bf16 hb = (__bf16)hv;
          *(__bf16*)&hbuf[(size_t)(row0 + ml) * H_ + n0 + wn + ns * 16 + (lane & 15)] = hb;
        }
      }
}

// ---------------- 7. down GEMM: y = h @ w_down^T (fp32 out) ---------------
// Block tile 128(M) x 128(N), 8 waves (4Mx2N), wave 2x4 tiles = 64 VGPR acc.
__global__ __launch_bounds__(256) void gemm_down(
    const unsigned short* __restrict__ hbuf, const float* __restrict__ wd,
    float* __restrict__ ybuf,
    const int* __restrict__ tileE, const int* __restrict__ tileRow0,
    const int* __restrict__ tileRows, const int* __restrict__ ntiles)
{
  __shared__ unsigned short As[2][128 * LDA];  // 20 KB
  __shared__ unsigned short Bs[2][128 * LDA];  // 20 KB

  int tile = blockIdx.y;
  if (tile >= ntiles[0]) return;
  int e = tileE[tile], row0 = tileRow0[tile], rows = tileRows[tile];
  int n0 = blockIdx.x * 128;

  int tid  = threadIdx.x;
  int lane = tid & 31, wid = tid >> 5;
  int wm = (wid & 3) * 32;
  int wn = (wid >> 2) * 64;
  const size_t wbase = (size_t)e * D_ * H_;

  fx8 cc[2][4];
  #pragma unroll
  for (int i = 0; i < 2; i++)
    #pragma unroll
    for (int j = 0; j < 4; j++) cc[i][j] = {};

  float4 bReg[4];

  auto asyncA = [&](int k0, int buf) {
    #pragma unroll
    for (int p = 0; p < 2; p++) {
      int c = tid + p * 256;
      int r = c >> 2, col = (c & 3) * 8;
      int rg = row0 + r; if (rg >= NT_) rg = NT_ - 1;
      unsigned voff = (unsigned)(((size_t)rg * H_ + k0 + col) * 2);
      async_copy_b128(hbuf, voff, &As[buf][r * LDA + col]);
    }
  };
  auto loadB = [&](int k0) {
    #pragma unroll
    for (int p = 0; p < 4; p++) {
      int c = tid + p * 256;
      int r = c >> 3, col = (c & 7) * 4;
      size_t off = wbase + (size_t)(n0 + r) * H_ + k0 + col;
      bReg[p] = *(const float4*)(wd + off);
      if (k0 + 2 * KT < H_)
        __builtin_prefetch(wd + off + 2 * KT, 0, 3);
    }
  };
  auto storeB = [&](int buf) {
    #pragma unroll
    for (int p = 0; p < 4; p++) {
      int c = tid + p * 256;
      int r = c >> 3, col = (c & 7) * 4;
      uint2 v;
      v.x = pack2bf(bReg[p].x, bReg[p].y);
      v.y = pack2bf(bReg[p].z, bReg[p].w);
      *(uint2*)&Bs[buf][r * LDA + col] = v;
    }
  };
  auto doMath = [&](int buf) {
    int kb = (lane < 16) ? 0 : 8;
    bfx16 aFrag[2];
    #pragma unroll
    for (int ms = 0; ms < 2; ms++) {
      int ar = (wm + ms * 16 + (lane & 15)) * LDA;
      bfx8 lo = *(const bfx8*)&As[buf][ar + kb];
      bfx8 hi = *(const bfx8*)&As[buf][ar + kb + 16];
      aFrag[ms] = CAT16(lo, hi);
    }
    #pragma unroll
    for (int ns = 0; ns < 4; ns++) {
      int br = (wn + ns * 16 + (lane & 15)) * LDA;
      bfx8 lo = *(const bfx8*)&Bs[buf][br + kb];
      bfx8 hi = *(const bfx8*)&Bs[buf][br + kb + 16];
      bfx16 bF = CAT16(lo, hi);
      #pragma unroll
      for (int ms = 0; ms < 2; ms++)
        cc[ms][ns] = __builtin_amdgcn_wmma_f32_16x16x32_bf16(
            false, aFrag[ms], false, bF, (short)0, cc[ms][ns], false, false);
    }
  };

  asyncA(0, 0);
  loadB(0);
  storeB(0);
  async_wait0();
  __syncthreads();
  const int NIT = H_ / KT;                     // 88
  #pragma unroll 2
  for (int kt = 0; kt < NIT; kt++) {
    int buf = kt & 1;
    if (kt + 1 < NIT) { asyncA((kt + 1) * KT, buf ^ 1); loadB((kt + 1) * KT); }
    doMath(buf);
    if (kt + 1 < NIT) storeB(buf ^ 1);
    async_wait0();
    __syncthreads();
  }

  int lhi = (lane >> 4) << 3;
  #pragma unroll
  for (int ms = 0; ms < 2; ms++)
    #pragma unroll
    for (int ns = 0; ns < 4; ns++)
      #pragma unroll
      for (int r = 0; r < 8; r++) {
        int ml = wm + ms * 16 + r + lhi;
        if (ml < rows)
          ybuf[(size_t)(row0 + ml) * D_ + n0 + wn + ns * 16 + (lane & 15)] = cc[ms][ns][r];
      }
}

// ---------------- 8. combine: out[s] = w0*y[p0] + w1*y[p1] ----------------
__global__ __launch_bounds__(256) void combine_kernel(
    const float* __restrict__ y, const float* __restrict__ rw,
    const int* __restrict__ pos_of_flat, float* __restrict__ out)
{
  int s = blockIdx.x;
  int p0 = pos_of_flat[s * 2], p1 = pos_of_flat[s * 2 + 1];
  float w0 = rw[s * 2], w1 = rw[s * 2 + 1];
  const float4* y0 = (const float4*)(y + (size_t)p0 * D_);
  const float4* y1 = (const float4*)(y + (size_t)p1 * D_);
  float4* o = (float4*)(out + (size_t)s * D_);
  int t = threadIdx.x;                         // 256 threads x float4 = 1024
  float4 a = y0[t], b = y1[t];
  float4 r;
  r.x = w0 * a.x + w1 * b.x;
  r.y = w0 * a.y + w1 * b.y;
  r.z = w0 * a.z + w1 * b.z;
  r.w = w0 * a.w + w1 * b.w;
  o[t] = r;
}

// ---------------- host launcher ----------------
extern "C" void kernel_launch(void* const* d_in, const int* in_sizes, int n_in,
                              void* d_out, int out_size, void* d_ws, size_t ws_size,
                              hipStream_t stream)
{
  const float* hidden = (const float*)d_in[0];
  const float* gate_w = (const float*)d_in[1];
  const float* w_gate = (const float*)d_in[2];
  const float* w_up   = (const float*)d_in[3];
  const float* w_down = (const float*)d_in[4];

  float* out    = (float*)d_out;
  float* logits = out + (size_t)S_ * D_;
  float* topi   = logits + (size_t)S_ * E_;

  char* p = (char*)d_ws;
  unsigned short* xs   = (unsigned short*)p; p += (size_t)NT_ * D_ * 2;  // 16 MB
  unsigned short* hbuf = (unsigned short*)p; p += (size_t)NT_ * H_ * 2;  // 46 MB
  float* ybuf          = (float*)p;          p += (size_t)NT_ * D_ * 4;  // 33.5 MB
  float* rw            = (float*)p;          p += (size_t)NT_ * 4;
  int* flat_e          = (int*)p;            p += (size_t)NT_ * 4;
  int* pos_of_flat     = (int*)p;            p += (size_t)NT_ * 4;
  int* row_of_pos      = (int*)p;            p += (size_t)NT_ * 4;
  int* counts          = (int*)p;            p += 64;
  int* cursors         = (int*)p;            p += 64;
  int* ntiles          = (int*)p;            p += 64;
  int* tileE           = (int*)p;            p += MAXTILES * 8;
  int* tileRow0        = (int*)p;            p += MAXTILES * 8;
  int* tileRows        = (int*)p;            p += MAXTILES * 8;

  init_kernel<<<1, 32, 0, stream>>>(counts, cursors, ntiles);
  router_kernel<<<S_ / 8, 256, 0, stream>>>(hidden, gate_w, logits, topi, rw, flat_e, counts);
  scan_kernel<<<1, 1, 0, stream>>>(counts, cursors, tileE, tileRow0, tileRows, ntiles);
  scatter_kernel<<<NT_ / 256, 256, 0, stream>>>(flat_e, cursors, pos_of_flat, row_of_pos);
  gather_kernel<<<NT_, 128, 0, stream>>>(hidden, row_of_pos, xs);
  gemm_gateup<<<dim3(H_ / 64, MAXTILES), 256, 0, stream>>>(
      xs, w_gate, w_up, hbuf, tileE, tileRow0, tileRows, ntiles);
  gemm_down<<<dim3(D_ / 128, MAXTILES), 256, 0, stream>>>(
      hbuf, w_down, ybuf, tileE, tileRow0, tileRows, ntiles);
  combine_kernel<<<S_, 256, 0, stream>>>(ybuf, rw, pos_of_flat, out);
}
